// CondConv_81063212745529
// MI455X (gfx1250) — matrix operand order repaired and checked by reading, not compiled, not run on hardware
//
#include <hip/hip_runtime.h>

typedef unsigned short u16;
typedef unsigned int   u32;
typedef unsigned long long u64;
typedef __attribute__((ext_vector_type(16))) __bf16 bf16x16;
typedef __attribute__((ext_vector_type(8)))  float  f32x8;
typedef __attribute__((ext_vector_type(4)))  u32    u32x4;
typedef __attribute__((ext_vector_type(4)))  int    i32x4;
typedef __attribute__((ext_vector_type(8)))  int    i32x8;

#define B_     16
#define CIN    256
#define COUT   256
#define H_     56
#define W_     56
#define HW     3136
#define E_     8
#define KK     2304      // cin * 9
#define NCHUNK 72        // KK / 32
#define WP     58
#define PLANE_P 3364     // 58*58

#if defined(__has_builtin)
#if __has_builtin(__builtin_amdgcn_tensor_load_to_lds)
#define HAVE_TDM 1
#endif
#endif

__device__ __forceinline__ u16 f2bf(float f) {
    u32 u = __float_as_uint(f);
    u32 r = (u + 0x7FFFu + ((u >> 16) & 1u)) >> 16;   // round-to-nearest-even
    return (u16)r;
}

#if HAVE_TDM
// Issue a 2D TDM load: tile (tile_d0 x tile_d1) elements of 2-byte data from a
// row-major tensor (row stride stride0 elements) into LDS at lds_off (packed rows).
__device__ __forceinline__ void tdm_load_2d_b16(const void* gaddr, u32 lds_off,
                                                u32 tile_d0, u32 tile_d1,
                                                u32 tensor_d0, u32 tensor_d1,
                                                u32 stride0) {
    u64 ga = (u64)(size_t)gaddr;
    u32x4 g0;
    g0[0] = 1u;                                   // count=1 (valid user D#)
    g0[1] = lds_off;                              // lds_addr (bytes)
    g0[2] = (u32)ga;                              // global_addr[31:0]
    g0[3] = (u32)(ga >> 32) | 0x80000000u;        // global_addr[56:32] | type=2<<30
    i32x8 g1;
    g1[0] = (int)(1u << 16);                      // wg_mask=0, data_size=1 (2 bytes)
    g1[1] = (int)((tensor_d0 & 0xFFFFu) << 16);   // [63:48] tensor_dim0 lo16
    g1[2] = (int)((tensor_d0 >> 16) | ((tensor_d1 & 0xFFFFu) << 16));
    g1[3] = (int)((tensor_d1 >> 16) | (tile_d0 << 16));   // [127:112] tile_dim0
    g1[4] = (int)(tile_d1 & 0xFFFFu);             // [143:128] tile_dim1 (tile_dim2=0)
    g1[5] = (int)stride0;                         // [191:160] tensor_dim0_stride lo32
    g1[6] = 0;
    g1[7] = 0;
    i32x4 z4 = (i32x4)0;
#if __clang_major__ >= 23
    i32x8 z8 = (i32x8)0;
    __builtin_amdgcn_tensor_load_to_lds(g0, g1, z4, z4, z8, 0);
#else
    __builtin_amdgcn_tensor_load_to_lds(g0, g1, z4, z4, 0);
#endif
}
#endif

// ---------------- Kernel 1: global average pool (per b,c plane), float4 loads ----------------
__global__ void pool_kernel(const float* __restrict__ x, float* __restrict__ pooled) {
    __shared__ float red[256];
    int idx = blockIdx.x;                        // b*256 + c
    const float4* p4 = (const float4*)(x + (size_t)idx * HW);   // 784 vec4, 16B aligned
    float s = 0.f;
    for (int i = threadIdx.x; i < HW / 4; i += 256) {
        float4 v = p4[i];
        s += (v.x + v.y) + (v.z + v.w);
    }
    red[threadIdx.x] = s;
    __syncthreads();
    for (int off = 128; off > 0; off >>= 1) {
        if (threadIdx.x < off) red[threadIdx.x] += red[threadIdx.x + off];
        __syncthreads();
    }
    if (threadIdx.x == 0) pooled[idx] = red[0] * (1.0f / (float)HW);
}

// ---------------- Kernel 2: routing = sigmoid(pooled @ Wr^T + br) ----------------
__global__ void route_kernel(const float* __restrict__ pooled,
                             const float* __restrict__ rw,
                             const float* __restrict__ rb,
                             float* __restrict__ r) {
    int b = blockIdx.x, e = threadIdx.x;
    if (e < E_) {
        float acc = rb[e];
        const float* pb = pooled + b * CIN;
        const float* we = rw + e * CIN;
        for (int c = 0; c < CIN; ++c) acc += pb[c] * we[c];
        r[b * E_ + e] = 1.0f / (1.0f + __expf(-acc));
    }
}

// ---------------- Kernel 2b: fold BN into scale/shift ----------------
__global__ void bn_kernel(const float* __restrict__ g,  const float* __restrict__ be,
                          const float* __restrict__ mn, const float* __restrict__ vr,
                          float* __restrict__ scale, float* __restrict__ shift) {
    int c = threadIdx.x;
    float inv = g[c] * rsqrtf(vr[c] + 1e-5f);
    scale[c] = inv;
    shift[c] = be[c] - mn[c] * inv;
}

// ---------------- Kernel 3: x (fp32 NCHW) -> zero-padded bf16 (16,256,58,58) ----------------
__global__ void pad_bf16_kernel(const float* __restrict__ x, u16* __restrict__ xp) {
    int idx = blockIdx.x;                        // b*256 + c
    const float* src = x + (size_t)idx * HW;
    u16* dst = xp + (size_t)idx * PLANE_P;
    for (int j = threadIdx.x; j < PLANE_P; j += 256) {
        int hh = j / WP, ww = j - hh * WP;
        int h = hh - 1, w = ww - 1;
        float v = (h >= 0 && h < H_ && w >= 0 && w < W_) ? src[h * W_ + w] : 0.f;
        dst[j] = f2bf(v);
    }
}

// ---------------- Kernel 4: cw[b,co,k] = sum_e r[b,e] * W[e,co,k], packed bf16x2 stores ----
__global__ void combine_kernel(const float* __restrict__ r,
                               const float* __restrict__ kw,
                               u16* __restrict__ cw) {
    int idx = blockIdx.x;                        // b*256 + co
    int b = idx >> 8, co = idx & 255;
    float rr[E_];
#pragma unroll
    for (int e = 0; e < E_; ++e) rr[e] = r[b * E_ + e];
    u32* dst = (u32*)(cw + (size_t)idx * KK);    // 4B aligned (KK even)
    for (int j2 = threadIdx.x; j2 < KK / 2; j2 += 256) {
        float a0 = 0.f, a1 = 0.f;
#pragma unroll
        for (int e = 0; e < E_; ++e) {
            const float2 w2 = *(const float2*)(kw + (size_t)(e * COUT + co) * KK + 2 * j2);
            a0 += rr[e] * w2.x;
            a1 += rr[e] * w2.y;
        }
        dst[j2] = (u32)f2bf(a0) | ((u32)f2bf(a1) << 16);
    }
}

// ---------------- Kernel 5: implicit-GEMM conv via v_wmma_f32_16x16x32_bf16 ----------------
// Block tile: M=128 couts x (4 output rows x 16 output cols). 8 waves = 2(M) x 4(row).
// A tile (128 rows x 32 K, bf16) DMA'd into LDS per chunk by the Tensor Data Mover
// (wave 0 issues, TENSORcnt fenced before the block barrier), double-buffered.
// B tile staged via LDS transpose-on-write into the wave32 B-fragment layout.
__global__ __launch_bounds__(256, 2)
void condconv_wmma_kernel(const u16* __restrict__ xp,
                          const u16* __restrict__ cw,
                          const float* __restrict__ scale,
                          const float* __restrict__ shift,
                          float* __restrict__ out) {
    __shared__ u16   ldsA[2][128 * 32];          // 2 x 8 KB: A tile, row-major 64B rows
    __shared__ u16   ldsB[2][4 * 2 * 16 * 16];   // 2 x 4 KB: B fragments
    __shared__ float sSc[128], sSh[128];         // BN params for this block's couts

    int bid = blockIdx.x;
    int wt = bid & 3;  bid >>= 2;                // 4 col tiles (16 wide, last half-masked)
    int ht = bid % 14; bid /= 14;                // 14 row tiles (4 rows each)
    int mt = bid & 1;  int b = bid >> 1;         // 2 M tiles, 16 samples
    int h0 = ht * 4, w0 = wt * 16, mbase = mt * 128;

    int tid  = threadIdx.x;
    int lane = tid & 31, wid = tid >> 5;
    int mw   = wid >> 2, nw = wid & 3;           // wave -> (M half, output row)
    int half = lane >> 4, nn = lane & 15;

    const u16* xpB   = xp + (size_t)b * CIN * PLANE_P;
    const u16* cwBlk = cw + ((size_t)b * COUT + mbase) * KK;   // 128 rows x KK

    if (tid < 128) {
        sSc[tid] = scale[mbase + tid];
        sSh[tid] = shift[mbase + tid];
    }

    f32x8 acc[4];
#pragma unroll
    for (int s = 0; s < 4; ++s)
#pragma unroll
        for (int i = 0; i < 8; ++i) acc[s][i] = 0.f;

    // B staging: 256 threads -> (row 0..3, t 0..31, pixel-half 0..1), 8 pixels each
    int combo = tid >> 1;
    int pxh   = tid & 1;
    int srow  = combo >> 5;
    int t     = combo & 31;
    const int ldsoff = (((srow * 2 + (t >> 4)) * 16) + pxh * 8) * 16 + (t & 15);

    // incremental decomposition of k = chunk*32 + t into (ci, kh, kw): 32 = 3*9 + 5
    int ci = t / 9;
    int r9 = t - ci * 9;

    auto stageB = [&](u16* buf, int ci_, int r9_) {
        int kh  = r9_ / 3;
        int kwv = r9_ - kh * 3;
        const u16* src = xpB + ci_ * PLANE_P + (h0 + srow + kh) * WP + (w0 + kwv + pxh * 8);
        u16* dstp = buf + ldsoff;
#pragma unroll
        for (int p = 0; p < 8; ++p) dstp[p * 16] = src[p];
    };

#if HAVE_TDM
    const u32 ldsAoff[2] = { (u32)(size_t)&ldsA[0][0], (u32)(size_t)&ldsA[1][0] };
    auto stageA = [&](int buf, int kb) {
        if (wid == 0)
            tdm_load_2d_b16(cwBlk + kb, ldsAoff[buf], /*tile*/32, 128,
                            /*tensor*/KK, 128, /*stride0*/KK);
    };
    auto fenceA = [&]() { if (wid == 0) __builtin_amdgcn_s_wait_tensorcnt(0); };
#else
    // fallback: cooperative global->LDS copy, same layout (thread: 16B of one row)
    auto stageA = [&](int buf, int kb) {
        int row  = tid >> 1;
        int part = tid & 1;
        const uint4* src = (const uint4*)(cwBlk + (size_t)row * KK + kb + part * 16);
        *(uint4*)&ldsA[buf][row * 32 + part * 16] = *src;
    };
    auto fenceA = [&]() {};
#endif

    // prologue: stage chunk 0 (A via TDM, B via transpose-stores)
    stageA(0, 0);
    stageB(ldsB[0], ci, r9);
    fenceA();
    __syncthreads();

#pragma unroll 2
    for (int it = 0; it < NCHUNK; ++it) {
        const int cur = it & 1;

        // ---- stage next chunk into the other buffers (overlaps with WMMA below) ----
        if (it + 1 < NCHUNK) {
            ci += 3; r9 += 5;
            if (r9 >= 9) { r9 -= 9; ci += 1; }
            stageB(ldsB[cur ^ 1], ci, r9);
            stageA(cur ^ 1, (it + 1) << 5);
        }

        // ---- A fragments from LDS (64B rows; two ds_load_b128 per sub-tile) ----
        bf16x16 afrag[4];
#pragma unroll
        for (int s = 0; s < 4; ++s) {
            const u16* aptr = &ldsA[cur][(mw * 64 + s * 16 + nn) * 32 + half * 8];
            union { uint4 q[2]; bf16x16 v; } ua;
            ua.q[0] = *(const uint4*)(aptr);          // K +0..7   (upper lanes: +8..15)
            ua.q[1] = *(const uint4*)(aptr + 16);     // K +16..23 (upper lanes: +24..31)
            afrag[s] = ua.v;
        }

        // ---- B fragment: two ds_load_b128 per lane ----
        union { uint4 q[2]; bf16x16 v; } ub;
        const u16* bptr = &ldsB[cur][((nw * 2) * 16 + nn) * 16 + half * 8];
        ub.q[0] = *(const uint4*)(bptr);
        ub.q[1] = *(const uint4*)(bptr + 256);
        bf16x16 bfrag = ub.v;

#pragma unroll
        for (int s = 0; s < 4; ++s)
            acc[s] = __builtin_amdgcn_wmma_f32_16x16x32_bf16(
                false, afrag[s], false, bfrag, (short)0, acc[s], false, false);

        fenceA();          // wave 0: TDM for next chunk complete before barrier publishes it
        __syncthreads();   // also: reads of buf[cur] done before it is restaged
    }

    // ---- epilogue: BN + SiLU, fp32 store. C layout: M = vr + 8*half, N = nn ----
    int h = h0 + nw;
    int w = w0 + nn;
    bool wok = (w < W_);
    float* outB = out + ((size_t)b * COUT + mbase + mw * 64) * HW + h * W_ + w;
#pragma unroll
    for (int s = 0; s < 4; ++s) {
#pragma unroll
        for (int vr = 0; vr < 8; ++vr) {
            int corel = mw * 64 + s * 16 + half * 8 + vr;
            float y = acc[s][vr] * sSc[corel] + sSh[corel];
            float o = y / (1.0f + __expf(-y));        // SiLU
            if (wok) outB[(size_t)(s * 16 + half * 8 + vr) * HW] = o;
        }
    }
}

// ---------------- launcher ----------------
extern "C" void kernel_launch(void* const* d_in, const int* in_sizes, int n_in,
                              void* d_out, int out_size, void* d_ws, size_t ws_size,
                              hipStream_t stream) {
    const float* x  = (const float*)d_in[0];
    const float* rw = (const float*)d_in[1];
    const float* rb = (const float*)d_in[2];
    const float* kw = (const float*)d_in[3];
    const float* g  = (const float*)d_in[4];
    const float* be = (const float*)d_in[5];
    const float* mn = (const float*)d_in[6];
    const float* vv = (const float*)d_in[7];
    float* out = (float*)d_out;

    // workspace layout (bytes), ~46.5 MB total
    char* ws = (char*)d_ws;
    float* pooled = (float*)(ws);                       // 16 KB
    float* r      = (float*)(ws + 16384);               // 512 B
    float* scale  = (float*)(ws + 16896);               // 1 KB
    float* shift  = (float*)(ws + 17920);               // 1 KB
    u16*   xp     = (u16*)(ws + 18944);                 // 16*256*58*58*2 = 27,557,888 B
    u16*   cw     = (u16*)(ws + 18944 + 27557888);      // 16*256*2304*2 = 18,874,368 B

    pool_kernel    <<<B_ * CIN, 256, 0, stream>>>(x, pooled);
    route_kernel   <<<B_, 64, 0, stream>>>(pooled, rw, rb, r);
    bn_kernel      <<<1, COUT, 0, stream>>>(g, be, mn, vv, scale, shift);
    pad_bf16_kernel<<<B_ * CIN, 256, 0, stream>>>(x, xp);
    combine_kernel <<<B_ * COUT, 256, 0, stream>>>(r, kw, cw);
    condconv_wmma_kernel<<<B_ * 2 * 14 * 4, 256, 0, stream>>>(xp, cw, scale, shift, out);
}
